// RNNNSAgent_89756226552351
// MI455X (gfx1250) — matrix operand order repaired and checked by reading, not compiled, not run on hardware
//
#include <hip/hip_runtime.h>

// ---------------------------------------------------------------------------
// Fused per-agent RNN (fc1+ReLU -> GRUCell -> fc2) for MI455X (gfx1250).
// bf16 WMMA (v_wmma_f32_16x16x32_bf16) with fp32 accumulation.
// One workgroup = 1 agent x 128 batch rows; 8 waves, 16 rows per wave.
// Barrier-free: LDS is a per-wave staging buffer for A-fragment transposes.
// All global accesses: one hoisted base pointer + constant immediate offsets.
// Gate nonlinearities: v_exp_f32 / v_rcp_f32 / v_tanh_f32 (no div refinement).
// ---------------------------------------------------------------------------

typedef __attribute__((ext_vector_type(16))) __bf16 v16bf;
typedef __attribute__((ext_vector_type(4)))  __bf16 v4bf;
typedef __attribute__((ext_vector_type(8)))  float  v8f;

#define NAGENT 8
#define DIN    128
#define HID    64
#define NACT   16
#define NBATCH 16384
#define LDSW   136   // row stride (elements) for the 16x128 bf16 tile (+8 pad)

// workspace offsets (bf16 elements)
#define OFF_W1   0
#define OFF_WIH  65536      // 8*64*128
#define OFF_WHH  163840     // + 8*192*64
#define OFF_W2   262144     // + 8*192*64
#define NW_TOT   270336     // + 8*16*64

__device__ __forceinline__ v16bf ld_frag2(const __bf16* p0, const __bf16* p1) {
  union { v16bf v; uint4 q[2]; } u;
  u.q[0] = *(const uint4*)p0;
  u.q[1] = *(const uint4*)p1;
  return u.v;
}
// B fragment: two 8-bf16 chunks (lanes 0-15 K=0..7,16..23; 16-31 K=8..15,24..31)
__device__ __forceinline__ v16bf ld_bfrag(const __bf16* p) { return ld_frag2(p, p + 16); }

__device__ __forceinline__ v8f wmma_bf16(v16bf a, v16bf b, v8f c) {
  return __builtin_amdgcn_wmma_f32_16x16x32_bf16(false, a, false, b, (short)0, c,
                                                 false, false);
}

__device__ __forceinline__ float fast_rcp(float x) {
#if __has_builtin(__builtin_amdgcn_rcpf)
  return __builtin_amdgcn_rcpf(x);      // single v_rcp_f32
#else
  return 1.0f / x;
#endif
}

__device__ __forceinline__ float sigmoid_f(float x) {
  return fast_rcp(1.0f + __expf(-x));   // v_exp_f32 + v_rcp_f32
}

__device__ __forceinline__ float tanh_f(float x) {
#if __has_builtin(__builtin_amdgcn_tanhf)
  return __builtin_amdgcn_tanhf(x);     // hardware v_tanh_f32
#else
  float e = __expf(-2.0f * x);
  return (1.0f - e) * fast_rcp(1.0f + e);
#endif
}

// --- weight fp32 -> bf16 conversion into workspace ---
__global__ void cvt_weights(const float* __restrict__ W1,
                            const float* __restrict__ Wih,
                            const float* __restrict__ Whh,
                            const float* __restrict__ W2,
                            __bf16* __restrict__ ws) {
  int i = blockIdx.x * 256 + threadIdx.x;
  if (i < OFF_WIH)        ws[i] = (__bf16)W1[i];
  else if (i < OFF_W2)    ws[i] = (i < OFF_WHH) ? (__bf16)Wih[i - OFF_WIH]
                                                : (__bf16)Whh[i - OFF_WHH];
  else if (i < NW_TOT)    ws[i] = (__bf16)W2[i - OFF_W2];
}

__global__ __launch_bounds__(256) void rnn_agent_fused(
    const float* __restrict__ x,    // [B*A, 128]
    const float* __restrict__ h0,   // [B, A, 64]
    const float* __restrict__ b1,   // [A, 64]
    const float* __restrict__ bih,  // [A, 192]
    const float* __restrict__ bhh,  // [A, 192]
    const float* __restrict__ b2,   // [A, 16]
    const __bf16* __restrict__ wbf, // workspace bf16 weights
    float* __restrict__ qout,       // [B*A, 16]
    float* __restrict__ hout)       // [B, A, 64]
{
  __shared__ __bf16 lds[8 * 16 * LDSW];

  const int tid  = threadIdx.x;
  const int lane = tid & 31;
  const int wave = tid >> 5;
  const int l16  = lane & 15;
  const int hi   = lane >> 4;          // lane-half select
  const int c0   = hi * 8;             // A-frag K sub-offset (elements)
  const int khi  = hi * 8;             // B-frag K sub-offset (elements)
  const int a    = blockIdx.x & (NAGENT - 1);
  const int bBase = (blockIdx.x >> 3) * 128 + wave * 16;  // first batch row

  __bf16* T = lds + wave * 16 * LDSW;  // this wave's 16 x 128 staging tile

  // ---- hoisted base pointers: all loop-varying terms become imm offsets ----
  const __bf16* w1p  = wbf + OFF_W1  + a * (HID * DIN) + l16 * DIN + khi; // + nt*16*DIN + ks*32
  const __bf16* wihp = wbf + OFF_WIH + a * (192 * HID) + l16 * HID + khi; // + row*HID + ks*32
  const __bf16* whhp = wbf + OFF_WHH + a * (192 * HID) + l16 * HID + khi;
  const __bf16* w2p  = wbf + OFF_W2  + a * (NACT * HID) + l16 * HID + khi; // + ks*32
  const float* xp  = x + ((size_t)bBase * NAGENT + a) * DIN + lane * 4;          // + r*(A*DIN)
  const float* hsp = h0 + ((size_t)(bBase + hi) * NAGENT + a) * HID + l16 * 4;   // + i*2*(A*HID)
  const float* hcp = h0   + ((size_t)(bBase + hi * 8) * NAGENT + a) * HID + l16; // + v*(A*HID)+j*16
  float*       hop = hout + ((size_t)(bBase + hi * 8) * NAGENT + a) * HID + l16;
  float*       qop = qout + ((size_t)(bBase + hi * 8) * NAGENT + a) * NACT + l16;
  const float* b1p  = b1  + a * HID + l16;   // + nt*16
  const float* bihp = bih + a * 192 + l16;   // + j*16 (+64 / +128)
  const float* bhhp = bhh + a * 192 + l16;
  const __bf16* ap0 = T + l16 * LDSW + c0;          // A-frags: + ks*32 (+16)
  __bf16*       tcw = T + hi * 8 * LDSW + l16;      // C-layout writes: + v*LDSW + col
  __bf16*       txw = T + lane * 4;                 // x staging: + r*LDSW
  __bf16*       thw = T + hi * LDSW + 64 + l16 * 4; // h staging: + i*2*LDSW

  // ---- stage: x (16 x 128 fp32) -> bf16 LDS ----
#pragma unroll
  for (int r = 0; r < 16; ++r) {
    const float4 f = *(const float4*)(xp + r * (NAGENT * DIN));
    v4bf t = { (__bf16)f.x, (__bf16)f.y, (__bf16)f.z, (__bf16)f.w };
    *(v4bf*)(txw + r * LDSW) = t;
  }

  // ---- GEMM1: z1 = relu(x @ W1^T + b1)   [16 x 64], K = 128 ----
  v8f acc1[4] = { v8f{}, v8f{}, v8f{}, v8f{} };
#pragma unroll
  for (int ks = 0; ks < 4; ++ks) {
    const __bf16* ap = ap0 + ks * 32;
    v16bf af = ld_frag2(ap, ap + 16);
#pragma unroll
    for (int nt = 0; nt < 4; ++nt)
      acc1[nt] = wmma_bf16(af, ld_bfrag(w1p + nt * 16 * DIN + ks * 32), acc1[nt]);
  }
  // bias + ReLU, write z1 bf16 into LDS cols 0..63 (x tile fully consumed)
#pragma unroll
  for (int nt = 0; nt < 4; ++nt) {
    const float bv = b1p[nt * 16];
#pragma unroll
    for (int v = 0; v < 8; ++v) {
      float z = fmaxf(acc1[nt][v] + bv, 0.0f);
      tcw[v * LDSW + nt * 16] = (__bf16)z;
    }
  }

  // ---- stage: h (16 x 64 fp32) -> bf16 LDS cols 64..127 ----
#pragma unroll
  for (int i = 0; i < 8; ++i) {
    const float4 f = *(const float4*)(hsp + i * 2 * (NAGENT * HID));
    v4bf t = { (__bf16)f.x, (__bf16)f.y, (__bf16)f.z, (__bf16)f.w };
    *(v4bf*)(thw + i * 2 * LDSW) = t;
  }

  // ---- hoist the four K=32 A-fragments of the [z1 | h] tile ----
  v16bf afr[4];
#pragma unroll
  for (int ks = 0; ks < 4; ++ks) {
    const __bf16* ap = ap0 + ks * 32;
    afr[ks] = ld_frag2(ap, ap + 16);
  }

  // ---- GEMM2 + GRU gates, per 16-col block j ----
  // r,z gates: fused K=128 over [z1 | h] against [W_ih ; W_hh] rows.
  // n gate: i_n (K=64, z1 x W_ih[128:192]) and h_n (K=64, h x W_hh[128:192]).
  float hnew[4][8];
#pragma unroll
  for (int j = 0; j < 4; ++j) {
    v8f ar = {}, az = {}, ain = {}, ahn = {};
#pragma unroll
    for (int ks = 0; ks < 4; ++ks) {
      const int rr = j * 16;         // r-gate weight row block
      const int rz = 64 + j * 16;    // z-gate weight row block
      const int rn = 128 + j * 16;   // n-gate weight row block
      const __bf16* wr = (ks < 2) ? (wihp + rr * HID + ks * 32)
                                  : (whhp + rr * HID + (ks - 2) * 32);
      const __bf16* wz = (ks < 2) ? (wihp + rz * HID + ks * 32)
                                  : (whhp + rz * HID + (ks - 2) * 32);
      ar = wmma_bf16(afr[ks], ld_bfrag(wr), ar);
      az = wmma_bf16(afr[ks], ld_bfrag(wz), az);
      if (ks < 2)
        ain = wmma_bf16(afr[ks], ld_bfrag(wihp + rn * HID + ks * 32), ain);
      else
        ahn = wmma_bf16(afr[ks], ld_bfrag(whhp + rn * HID + (ks - 2) * 32), ahn);
    }
    const float br  = bihp[j * 16] + bhhp[j * 16];
    const float bz  = bihp[64 + j * 16] + bhhp[64 + j * 16];
    const float bin = bihp[128 + j * 16];
    const float bhn = bhhp[128 + j * 16];
#pragma unroll
    for (int v = 0; v < 8; ++v) {
      const float r  = sigmoid_f(ar[v] + br);
      const float zg = sigmoid_f(az[v] + bz);
      const float ng = tanh_f((ain[v] + bin) + r * (ahn[v] + bhn));
      const float hv = hcp[v * (NAGENT * HID) + j * 16];   // fp32 h for the blend
      const float hn = ng + zg * (hv - ng);                // (1-z)*n + z*h
      hnew[j][v] = hn;
      hop[v * (NAGENT * HID) + j * 16] = hn;
    }
  }

  // ---- hnew -> bf16 LDS cols 0..63 (z1 region fully consumed) ----
#pragma unroll
  for (int j = 0; j < 4; ++j)
#pragma unroll
    for (int v = 0; v < 8; ++v)
      tcw[v * LDSW + j * 16] = (__bf16)hnew[j][v];

  // ---- GEMM3: q = hnew @ W2^T + b2   [16 x 16], K = 64 ----
  v8f aq = {};
#pragma unroll
  for (int ks = 0; ks < 2; ++ks) {
    const __bf16* ap = ap0 + ks * 32;
    aq = wmma_bf16(ld_frag2(ap, ap + 16), ld_bfrag(w2p + ks * 32), aq);
  }
  const float bq = b2[a * NACT + l16];
#pragma unroll
  for (int v = 0; v < 8; ++v)
    qop[v * (NAGENT * NACT)] = aq[v] + bq;
}

extern "C" void kernel_launch(void* const* d_in, const int* in_sizes, int n_in,
                              void* d_out, int out_size, void* d_ws, size_t ws_size,
                              hipStream_t stream) {
  const float* x   = (const float*)d_in[0];
  const float* h0  = (const float*)d_in[1];
  const float* W1  = (const float*)d_in[2];
  const float* b1  = (const float*)d_in[3];
  const float* Wih = (const float*)d_in[4];
  const float* bih = (const float*)d_in[5];
  const float* Whh = (const float*)d_in[6];
  const float* bhh = (const float*)d_in[7];
  const float* W2  = (const float*)d_in[8];
  const float* b2  = (const float*)d_in[9];

  float*  out = (float*)d_out;
  __bf16* ws  = (__bf16*)d_ws;   // 270336 bf16 = 528 KiB of scratch

  // q occupies B*A*NACT floats, h_new follows.
  float* qout = out;
  float* hout = out + (size_t)NBATCH * NAGENT * NACT;

  cvt_weights<<<NW_TOT / 256, 256, 0, stream>>>(W1, Wih, Whh, W2, ws);

  const int nblocks = (NBATCH / 128) * NAGENT;  // 1024
  rnn_agent_fused<<<nblocks, 256, 0, stream>>>(x, h0, b1, bih, bhh, b2, ws,
                                               qout, hout);
}
